// attention_24592982737269
// MI455X (gfx1250) — matrix operand order repaired and checked by reading, not compiled, not run on hardware
//
#include <hip/hip_runtime.h>
#include <math.h>

#define H 128
#define B 1024
#define S 1024
#define CHUNK 64            // enc rows staged in LDS per iteration
#define ROWU  68            // uints per padded LDS row: 64 data + 4 pad (272 B) -> conflict-free

typedef __attribute__((ext_vector_type(16))) __bf16 v16bf;
typedef __attribute__((ext_vector_type(8)))  float  v8f;

union Frag {
  unsigned int u[8];
  uint4 q[2];
  v16bf bf;
};

// round-to-nearest-even f32 -> bf16 (as uint16 in low bits)
__device__ __forceinline__ unsigned int f2bf(float f) {
  unsigned int u = __float_as_uint(f);
  return (u + 0x7FFFu + ((u >> 16) & 1u)) >> 16;
}
__device__ __forceinline__ unsigned int pack_bf2(float lo, float hi) {
  return f2bf(lo) | (f2bf(hi) << 16);
}

// gfx1250 hardware tanh (single trans-op, branchless)
__device__ __forceinline__ float fast_tanh(float x) {
#if __has_builtin(__builtin_amdgcn_tanhf)
  return __builtin_amdgcn_tanhf(x);
#else
  float r;
  asm("v_tanh_f32 %0, %1" : "=v"(r) : "v"(x));
  return r;
#endif
}

// x + dpp_permuted(x): VALU-only lane exchange, no LDS, no waits.
// DPP control must be an integer constant -> template parameter.
template <int CTRL>
__device__ __forceinline__ float dpp_add(float x) {
  int t = __builtin_amdgcn_update_dpp(0, __float_as_int(x), CTRL, 0xf, 0xf, true);
  return x + __int_as_float(t);
}
// sum across each group of 16 lanes (valid butterfly: quad xor1, quad xor2,
// half-mirror pairs quads, mirror pairs octets)
__device__ __forceinline__ float row16_sum(float x) {
  x = dpp_add<0x0B1>(x);   // quad_perm [1,0,3,2]
  x = dpp_add<0x04E>(x);   // quad_perm [2,3,0,1]
  x = dpp_add<0x141>(x);   // row_half_mirror
  x = dpp_add<0x140>(x);   // row_mirror
  return x;
}

// ---------------- Phase 1: ehpb[b,o] = hidden[b,:].Wh[o,:] + bias[o] ----------------
__global__ void eh_kernel(const float* __restrict__ hidden,
                          const float* __restrict__ W,
                          const float* __restrict__ bias,
                          float* __restrict__ ehpb) {
  __shared__ float s_h[H];
  const int b = blockIdx.x, o = threadIdx.x;
  s_h[o] = hidden[b * H + o];
  __syncthreads();
  const float4* wr = (const float4*)(W + (size_t)o * (2 * H)); // Wh[o,h] = W[o*256 + h]
  float acc = bias[o];
#pragma unroll 8
  for (int i = 0; i < H / 4; ++i) {
    float4 w4 = wr[i];
    acc += s_h[4*i+0]*w4.x + s_h[4*i+1]*w4.y + s_h[4*i+2]*w4.z + s_h[4*i+3]*w4.w;
  }
  ehpb[b * H + o] = acc;
}

// ---------------- Phase 2: fused ee GEMM + tanh + v-dot + softmax ----------------
__global__ void __launch_bounds__(256) attn_kernel(
    const float* __restrict__ enc, const float* __restrict__ W,
    const float* __restrict__ v,   const float* __restrict__ ehpb,
    float* __restrict__ out) {
  __shared__ unsigned int s_a[CHUNK * ROWU];   // bf16 A tiles (padded rows)
  __shared__ float s_part[8][S];               // per-wave score partitions (no atomics)
  __shared__ float s_red[16];

  const int tid  = threadIdx.x;
  const int wid  = tid >> 5;        // wave id 0..7 -> owns o-tile
  const int lane = tid & 31;
  const int hs   = lane >> 4;       // half-wave select
  const int n    = lane & 15;       // N column within 16x16 tile
  const int b    = blockIdx.x;
  const int o    = wid * 16 + n;    // output feature owned by this lane

  // Persistent B fragments: B[k,n] = We[o,k] = W[o*2H + H + k], bf16, K=128 in 4 chunks.
  Frag bfrag[4];
  {
    const float* wrow = W + (size_t)o * (2 * H) + H;
#pragma unroll
    for (int kc = 0; kc < 4; ++kc) {
      const float4* p = (const float4*)(wrow + kc * 32 + hs * 8);
#pragma unroll
      for (int j = 0; j < 4; ++j) {
        float4 f = p[j + (j & 2)];
        bfrag[kc].u[2*j+0] = pack_bf2(f.x, f.y);
        bfrag[kc].u[2*j+1] = pack_bf2(f.z, f.w);
      }
    }
  }
  const float ehv = ehpb[b * H + o];   // eh[b,o] + bias[o]
  const float vv  = v[o];
  const float* encb = enc + (size_t)b * S * H;

  for (int ch = 0; ch < S / CHUNK; ++ch) {
    __syncthreads();   // previous chunk's A reads complete before overwrite
    // Stage CHUNK x H f32 -> bf16 into LDS; 4 threads per row, coalesced b128 loads.
    {
      const int r = tid >> 2, q = tid & 3;
      const float4* src = (const float4*)(encb + (size_t)(ch * CHUNK + r) * H + q * 32);
      unsigned int* dst = s_a + r * ROWU + q * 16;
#pragma unroll
      for (int j = 0; j < 8; ++j) {
        float4 f = src[j];
        ((uint2*)dst)[j] = make_uint2(pack_bf2(f.x, f.y), pack_bf2(f.z, f.w));
      }
      if (ch + 1 < S / CHUNK)
        __builtin_prefetch((const char*)src + (size_t)CHUNK * H * sizeof(float), 0, 1);
    }
    __syncthreads();

    // Each wave: all 4 s-tiles of this chunk against its persistent B fragments.
#pragma unroll 1
    for (int t = 0; t < CHUNK / 16; ++t) {
      const unsigned int* p = s_a + (t * 16 + n) * ROWU + hs * 4;
      Frag a0, a1, a2, a3;
      a0.q[0] = *(const uint4*)(p +  0); a0.q[1] = *(const uint4*)(p +  8);
      a1.q[0] = *(const uint4*)(p + 16); a1.q[1] = *(const uint4*)(p + 24);
      a2.q[0] = *(const uint4*)(p + 32); a2.q[1] = *(const uint4*)(p + 40);
      a3.q[0] = *(const uint4*)(p + 48); a3.q[1] = *(const uint4*)(p + 56);
      v8f c = {0.f, 0.f, 0.f, 0.f, 0.f, 0.f, 0.f, 0.f};
      c = __builtin_amdgcn_wmma_f32_16x16x32_bf16(false, a0.bf, false, bfrag[0].bf,
                                                  (short)0, c, false, false);
      c = __builtin_amdgcn_wmma_f32_16x16x32_bf16(false, a1.bf, false, bfrag[1].bf,
                                                  (short)0, c, false, false);
      c = __builtin_amdgcn_wmma_f32_16x16x32_bf16(false, a2.bf, false, bfrag[2].bf,
                                                  (short)0, c, false, false);
      c = __builtin_amdgcn_wmma_f32_16x16x32_bf16(false, a3.bf, false, bfrag[3].bf,
                                                  (short)0, c, false, false);
      // scores[s] = sum_o v[o]*tanh(ee + eh + bias); C layout: c[r] = D[M=r+8*hs, N=n].
      // DPP butterfly gives the 16-lane sum in ALL lanes; lane n keeps r==n's value,
      // then lanes n<8 store the tile's 16 scores with one plain LDS store.
      float keep = 0.f;
#pragma unroll
      for (int r = 0; r < 8; ++r) {
        float val = row16_sum(vv * fast_tanh(c[r] + ehv));
        keep = (n == r) ? val : keep;
      }
      if (n < 8)
        s_part[wid][ch * CHUNK + t * 16 + hs * 8 + n] = keep;
    }
  }
  __syncthreads();

  // ---------------- softmax over S within the block ----------------
  float sc[4], ex[4];
  float m = -3.0e38f;
#pragma unroll
  for (int i = 0; i < 4; ++i) {
    const int idx = tid + 256 * i;
    float s = 0.f;
#pragma unroll
    for (int w = 0; w < 8; ++w) s += s_part[w][idx];
    sc[i] = s;
    m = fmaxf(m, s);
  }
#pragma unroll
  for (int msk = 16; msk >= 1; msk >>= 1) m = fmaxf(m, __shfl_xor(m, msk, 32));
  if (lane == 0) s_red[wid] = m;
  __syncthreads();
  float bmax = s_red[0];
#pragma unroll
  for (int j = 1; j < 8; ++j) bmax = fmaxf(bmax, s_red[j]);
  float sum = 0.f;
#pragma unroll
  for (int i = 0; i < 4; ++i) { ex[i] = __expf(sc[i] - bmax); sum += ex[i]; }
#pragma unroll
  for (int msk = 16; msk >= 1; msk >>= 1) sum += __shfl_xor(sum, msk, 32);
  if (lane == 0) s_red[8 + wid] = sum;
  __syncthreads();
  float tot = 0.f;
#pragma unroll
  for (int j = 0; j < 8; ++j) tot += s_red[8 + j];
  const float inv = 1.0f / tot;
#pragma unroll
  for (int i = 0; i < 4; ++i) out[(size_t)b * S + tid + 256 * i] = ex[i] * inv;
}

extern "C" void kernel_launch(void* const* d_in, const int* in_sizes, int n_in,
                              void* d_out, int out_size, void* d_ws, size_t ws_size,
                              hipStream_t stream) {
  (void)in_sizes; (void)n_in; (void)out_size; (void)ws_size;
  const float* hidden = (const float*)d_in[0];
  const float* enc    = (const float*)d_in[1];
  const float* W      = (const float*)d_in[2];
  const float* bias   = (const float*)d_in[3];
  const float* v      = (const float*)d_in[4];
  float* out  = (float*)d_out;
  float* ehpb = (float*)d_ws;                 // B*H f32 scratch (512 KB)

  eh_kernel<<<B, H, 0, stream>>>(hidden, W, bias, ehpb);
  attn_kernel<<<B, 256, 0, stream>>>(enc, W, v, ehpb, out);
}
